// SNN_50946902065743
// MI455X (gfx1250) — matrix operand order repaired and checked by reading, not compiled, not run on hardware
//
#include <hip/hip_runtime.h>

// SNN scan for MI455X (gfx1250, wave32, WMMA bf16 path).
// Hoisted input GEMM (parallel over all T, 32x32 register-blocked) + per-step
// recurrent WMMA GEMM with fused LIF dynamics + WMMA readout.
// w_rec (2MB bf16) stays L2-resident across the 100 serial steps.

#define T_STEPS 100
#define BATCH   256
#define HID     1024
#define NIN     784
#define NIN_PAD 800      // 25 * 32
#define NOUT    10
#define NOUT_PAD 16

typedef __attribute__((ext_vector_type(16))) __bf16 v16bf;
typedef __attribute__((ext_vector_type(8)))  __bf16 v8bf;
typedef __attribute__((ext_vector_type(8)))  float  v8f;

union ABfrag { v16bf v; v8bf h[2]; };

__device__ __forceinline__ unsigned short f32_to_bf16_rne(float f) {
  unsigned int u = __float_as_uint(f);
  u += 0x7FFFu + ((u >> 16) & 1u);
  return (unsigned short)(u >> 16);
}

__device__ __forceinline__ v8f wmma_bf16(v16bf a, v16bf b, v8f c) {
  return __builtin_amdgcn_wmma_f32_16x16x32_bf16(false, a, false, b,
                                                 (short)0, c, false, false);
}

// A-matrix 16x32 bf16 fragment (ISA 7.12.2): lane 0-15 row M=lane, K={0..7,16..23};
// lanes 16-31 K={8..15,24..31}. Two 16-byte chunks per lane.
__device__ __forceinline__ v16bf load_a(const unsigned short* __restrict__ base,
                                        int ld, int mbase, int k, int lane) {
  int row = mbase + (lane & 15);
  int hi  = (lane >> 4) << 3;                 // 0 or 8
  const unsigned short* p = base + (long long)row * ld + k + hi;
  ABfrag f;
  f.h[0] = *reinterpret_cast<const v8bf*>(p);
  f.h[1] = *reinterpret_cast<const v8bf*>(p + 16);
  return f.v;
}

// B-matrix 32x16 bf16 fragment: lane holds output column n = lane&15;
// lanes 0-15 K={0..7,16..23}, lanes 16-31 K={8..15,24..31} (matches A fragment).
__device__ __forceinline__ v16bf load_b(const unsigned short* __restrict__ base,
                                        int ld, int nbase, int k, int lane) {
  int row = nbase + (lane & 15);              // weight row == output column
  int hi  = (lane >> 4) << 3;                 // 0 or 8
  const unsigned short* p = base + (long long)row * ld + k + hi;
  ABfrag f;
  f.h[0] = *reinterpret_cast<const v8bf*>(p);
  f.h[1] = *reinterpret_cast<const v8bf*>(p + 16);
  return f.v;
}

// ---------------- conversion / init ----------------

__global__ __launch_bounds__(256) void k_cvt_pad(const float* __restrict__ src,
                                                 unsigned short* __restrict__ dst,
                                                 int dstRows, int dstK,
                                                 int srcRows, int srcK) {
  long long idx = (long long)blockIdx.x * blockDim.x + threadIdx.x;
  long long total = (long long)dstRows * dstK;
  if (idx >= total) return;
  int r = (int)(idx / dstK);
  int c = (int)(idx % dstK);
  float v = (r < srcRows && c < srcK) ? src[(long long)r * srcK + c] : 0.0f;
  dst[idx] = f32_to_bf16_rne(v);
}

__global__ __launch_bounds__(256) void k_zero(unsigned int* __restrict__ p, long long nWords) {
  long long idx = (long long)blockIdx.x * blockDim.x + threadIdx.x;
  if (idx < nWords) p[idx] = 0u;
}

// ---------------- hoisted input GEMM: Cin = Xb @ Winb^T ----------------
// M = T*B = 25600, N = HID = 1024, K = 800.
// 32x32 register-blocked tile per wave: 4 WMMAs per 8 b128 loads (2x better
// arithmetic intensity than 16x16), halves redundant A/B panel traffic.
__global__ __launch_bounds__(256) void k_gemm_in(const unsigned short* __restrict__ X,
                                                 const unsigned short* __restrict__ W,
                                                 float* __restrict__ C) {
  int wid  = blockIdx.x * 8 + (threadIdx.x >> 5);
  int lane = threadIdx.x & 31;
  const int Ntiles = HID / 32;                // 32
  int mbase = (wid / Ntiles) * 32;
  int nbase = (wid % Ntiles) * 32;
  v8f acc00 = {}, acc01 = {}, acc10 = {}, acc11 = {};
  for (int k = 0; k < NIN_PAD; k += 32) {
    v16bf a0 = load_a(X, NIN_PAD, mbase,      k, lane);
    v16bf a1 = load_a(X, NIN_PAD, mbase + 16, k, lane);
    v16bf b0 = load_b(W, NIN_PAD, nbase,      k, lane);
    v16bf b1 = load_b(W, NIN_PAD, nbase + 16, k, lane);
    acc00 = wmma_bf16(a0, b0, acc00);
    acc01 = wmma_bf16(a0, b1, acc01);
    acc10 = wmma_bf16(a1, b0, acc10);
    acc11 = wmma_bf16(a1, b1, acc11);
  }
  int col   = nbase + (lane & 15);
  int rbase = mbase + ((lane >> 4) << 3);
#pragma unroll
  for (int r = 0; r < 8; ++r) {
    long long row0 = rbase + r;
    C[row0 * HID + col]               = acc00[r];
    C[row0 * HID + col + 16]          = acc01[r];
    C[(row0 + 16) * HID + col]        = acc10[r];
    C[(row0 + 16) * HID + col + 16]   = acc11[r];
  }
}

// ---------------- per-step: recurrent GEMM + fused LIF update ----------------
// i_new = 0.8*i + cin_t + z_old @ Wrec^T ; v_dec = 0.9*v + 0.1*i ;
// z_new = (v_dec - 0.5 > 0) ; v_new = (1-z_new)*v_dec
// 16x16 tiles (1024 waves): the serial chain is latency-bound and all operands
// are L2-resident, so maximize parallelism rather than register blocking.
__global__ __launch_bounds__(256) void k_step(const unsigned short* __restrict__ zin,
                                              unsigned short* __restrict__ zout,
                                              float* __restrict__ V,
                                              float* __restrict__ I,
                                              const float* __restrict__ cin_t,
                                              const unsigned short* __restrict__ Wrec) {
  int wid  = blockIdx.x * 8 + (threadIdx.x >> 5);   // 0..1023
  int lane = threadIdx.x & 31;
  int nbase = (wid >> 4) * 16;      // 16 m-tiles share an n-tile -> Wrec rows reused
  int mbase = (wid & 15) * 16;
  v8f acc = {};
  for (int k = 0; k < HID; k += 32) {
    v16bf a = load_a(zin,  HID, mbase, k, lane);
    v16bf b = load_b(Wrec, HID, nbase, k, lane);
    acc = wmma_bf16(a, b, acc);
  }
  int col   = nbase + (lane & 15);
  int rbase = mbase + ((lane >> 4) << 3);
#pragma unroll
  for (int r = 0; r < 8; ++r) {
    long long idx = (long long)(rbase + r) * HID + col;
    float vold = V[idx];
    float iold = I[idx];
    float vdec = 0.9f * vold + 0.1f * iold;           // 1-DT*TAU_MEM_INV, DT*TAU_MEM_INV
    float z    = (vdec - 0.5f) > 0.0f ? 1.0f : 0.0f;  // V_TH
    V[idx]    = (1.0f - z) * vdec;
    I[idx]    = 0.8f * iold + cin_t[idx] + acc[r];    // 1-DT*TAU_SYN_INV
    zout[idx] = (z != 0.0f) ? (unsigned short)0x3F80u : (unsigned short)0u; // bf16 1.0/0.0
  }
}

// ---------------- per-step readout: y = z_new @ Wout^T, LI cell, emit vo ----------------
__global__ __launch_bounds__(256) void k_readout(const unsigned short* __restrict__ z,
                                                 const unsigned short* __restrict__ Wout,
                                                 float* __restrict__ Vo,
                                                 float* __restrict__ Io,
                                                 float* __restrict__ out_t) {
  int wid  = blockIdx.x * 8 + (threadIdx.x >> 5);   // 0..15 -> m-tile over batch
  int lane = threadIdx.x & 31;
  int mbase = wid * 16;
  v8f acc = {};
  for (int k = 0; k < HID; k += 32) {
    v16bf a = load_a(z,    HID, mbase, k, lane);
    v16bf b = load_b(Wout, HID, 0,     k, lane);    // rows 10..15 are zero padding
    acc = wmma_bf16(a, b, acc);
  }
  int o     = lane & 15;
  int rbase = mbase + ((lane >> 4) << 3);
#pragma unroll
  for (int r = 0; r < 8; ++r) {
    int b = rbase + r;
    int sidx = b * NOUT_PAD + o;
    float voold = Vo[sidx];
    float ioold = Io[sidx];
    float vonew = 0.9f * voold + 0.1f * ioold;   // LI cell on OLD io (reference order)
    Vo[sidx] = vonew;
    Io[sidx] = 0.8f * ioold + acc[r];
    if (o < NOUT) out_t[b * NOUT + o] = vonew;
  }
}

// ---------------- host-side orchestration ----------------

extern "C" void kernel_launch(void* const* d_in, const int* in_sizes, int n_in,
                              void* d_out, int out_size, void* d_ws, size_t ws_size,
                              hipStream_t stream) {
  (void)in_sizes; (void)n_in; (void)out_size; (void)ws_size;
  const float* x     = (const float*)d_in[0];   // [100,256,1,28,28]
  const float* w_in  = (const float*)d_in[1];   // [1024,784]
  const float* w_rec = (const float*)d_in[2];   // [1024,1024]
  const float* w_out = (const float*)d_in[3];   // [10,1024]
  float* out = (float*)d_out;                   // [100,256,10]

  char* ws = (char*)d_ws;
  size_t off = 0;
  auto wsalloc = [&](size_t bytes) -> void* {
    void* p = ws + off;
    off += (bytes + 255) & ~(size_t)255;
    return p;
  };
  unsigned short* Xb    = (unsigned short*)wsalloc((size_t)T_STEPS * BATCH * NIN_PAD * 2);
  unsigned short* Winb  = (unsigned short*)wsalloc((size_t)HID * NIN_PAD * 2);
  unsigned short* Wrecb = (unsigned short*)wsalloc((size_t)HID * HID * 2);
  unsigned short* Woutb = (unsigned short*)wsalloc((size_t)NOUT_PAD * HID * 2);
  float*          Cin   = (float*)wsalloc((size_t)T_STEPS * BATCH * HID * 4);
  unsigned short* Z0    = (unsigned short*)wsalloc((size_t)BATCH * HID * 2);
  unsigned short* Z1    = (unsigned short*)wsalloc((size_t)BATCH * HID * 2);
  float*          V     = (float*)wsalloc((size_t)BATCH * HID * 4);
  float*          I     = (float*)wsalloc((size_t)BATCH * HID * 4);
  float*          Vo    = (float*)wsalloc((size_t)BATCH * NOUT_PAD * 4);
  float*          Io    = (float*)wsalloc((size_t)BATCH * NOUT_PAD * 4);
  char*           wsEnd = ws + off;

  // 1) convert/pad weights & inputs to bf16 (every call; deterministic)
  {
    long long n;
    n = (long long)T_STEPS * BATCH * NIN_PAD;
    k_cvt_pad<<<(unsigned)((n + 255) / 256), 256, 0, stream>>>(
        x, Xb, T_STEPS * BATCH, NIN_PAD, T_STEPS * BATCH, NIN);
    n = (long long)HID * NIN_PAD;
    k_cvt_pad<<<(unsigned)((n + 255) / 256), 256, 0, stream>>>(
        w_in, Winb, HID, NIN_PAD, HID, NIN);
    n = (long long)HID * HID;
    k_cvt_pad<<<(unsigned)((n + 255) / 256), 256, 0, stream>>>(
        w_rec, Wrecb, HID, HID, HID, HID);
    n = (long long)NOUT_PAD * HID;
    k_cvt_pad<<<(unsigned)((n + 255) / 256), 256, 0, stream>>>(
        w_out, Woutb, NOUT_PAD, HID, NOUT, HID);
  }

  // 2) zero-init state region (Z0..Io are contiguous in ws)
  {
    long long nWords = (long long)(wsEnd - (char*)Z0) / 4;
    k_zero<<<(unsigned)((nWords + 255) / 256), 256, 0, stream>>>((unsigned int*)Z0, nWords);
  }

  // 3) hoisted input GEMM: Cin[t*B+b, h] = x_t @ w_in^T  (41 GFLOP, fully parallel)
  {
    const int waves = (T_STEPS * BATCH / 32) * (HID / 32);   // 25600 waves
    k_gemm_in<<<waves / 8, 256, 0, stream>>>(Xb, Winb, Cin);
  }

  // 4) serial scan: recurrent WMMA GEMM + LIF, then WMMA readout, per step
  const unsigned short* zin = Z0;
  unsigned short*       zout = Z1;
  for (int t = 0; t < T_STEPS; ++t) {
    const float* cin_t = Cin + (long long)t * BATCH * HID;
    k_step<<<(BATCH / 16) * (HID / 16) / 8, 256, 0, stream>>>(
        zin, zout, V, I, cin_t, Wrecb);
    k_readout<<<2, 256, 0, stream>>>(
        zout, Woutb, Vo, Io, out + (long long)t * BATCH * NOUT);
    const unsigned short* tmp = zout;
    zout = (unsigned short*)zin;
    zin  = tmp;
  }
}
